// SpatialMHSA_10428180594877
// MI455X (gfx1250) — matrix operand order repaired and checked by reading, not compiled, not run on hardware
//
#include <hip/hip_runtime.h>

typedef __attribute__((ext_vector_type(16))) _Float16 v16h;
typedef __attribute__((ext_vector_type(8)))  float    v8f;

#define WMMA_F16(a, b, c) \
  __builtin_amdgcn_wmma_f32_16x16x32_f16(false, (a), false, (b), (short)0, (c), false, false)

constexpr int NB = 2;      // batch
constexpr int C  = 512;    // channels
constexpr int S  = 4096;   // tokens (64*64)
constexpr int NH = 8;      // heads
constexpr int HD = 64;     // head dim

// ---------------- async global->LDS copy (16 bytes per lane) ----------------

#if defined(__has_builtin)
#  if __has_builtin(__builtin_amdgcn_global_load_async_to_lds_b128)
#    define HAS_ASYNC_LDS 1
#  else
#    define HAS_ASYNC_LDS 0
#  endif
#else
#  define HAS_ASYNC_LDS 0
#endif

typedef int v4i_vs __attribute__((vector_size(16)));
typedef __attribute__((address_space(1))) v4i_vs* gv4i_p;   // global AS pointer to int4
typedef __attribute__((address_space(3))) v4i_vs* lv4i_p;   // LDS AS pointer to int4

__device__ __forceinline__ void cp16(const _Float16* g, _Float16* l) {
#if HAS_ASYNC_LDS
  __builtin_amdgcn_global_load_async_to_lds_b128(
      (gv4i_p)(unsigned long long)(size_t)g,
      (lv4i_p)(unsigned int)(size_t)l,
      0, 0);
#else
  *(float4*)l = *(const float4*)g;
#endif
}

__device__ __forceinline__ void wait_async0() {
#if HAS_ASYNC_LDS
  asm volatile("s_wait_asynccnt 0" ::: "memory");
#endif
}

// ---------------- WMMA fragment loaders (CDNA5 wave32; A and B both use contiguous k = 16*half + e) ----------------

// A-matrix 16x32 f16 from row-major [m][k] LDS tile.
__device__ __forceinline__ v16h load_a_frag(const _Float16* A, int lda, int lane) {
  const int m  = lane & 15;
  const int k0 = (lane >> 4) * 16;
  v16h a;
#pragma unroll
  for (int e = 0; e < 16; ++e) a[e] = A[m * lda + k0 + e];
  return a;
}

// B-matrix 32x16 (logical B[k][n]) stored transposed in LDS as T[n][k].
__device__ __forceinline__ v16h load_bt_frag(const _Float16* T, int ldt, int lane) {
  const int n  = lane & 15;
  const int k0 = (lane >> 4) * 16;
  v16h b;
#pragma unroll
  for (int e = 0; e < 16; ++e) b[e] = T[n * ldt + k0 + e];
  return b;
}

// ---------------- conversion kernels ----------------

__global__ void f32_to_f16_kernel(const float* __restrict__ in, _Float16* __restrict__ out, int n) {
  int i = blockIdx.x * blockDim.x + threadIdx.x;
  if (i < n) out[i] = (_Float16)in[i];
}

// x [NB][C][S] f32 -> Xt [NB][S][C] f16 (token-major)
__global__ void xpose_f16_kernel(const float* __restrict__ x, _Float16* __restrict__ Xt) {
  size_t i = (size_t)blockIdx.x * blockDim.x + threadIdx.x;  // over NB*C*S
  int s = (int)(i % S);
  size_t r = i / S;
  int c = (int)(r % C);
  int n = (int)(r / C);
  Xt[((size_t)n * S + s) * C + c] = (_Float16)x[i];
}

// ---------------- QKV projection: Out = W (512x512) * X^T + b ----------------
// grid (C/64, NB*S/256, 3), block 256; 8 waves 2x4; wave tile 32x64 (8 WMMA / K-step)
// Q,K stored token-major [n][h][s][d]; V stored channel-major [n][c][s].

__global__ __launch_bounds__(256)
void qkv_gemm_kernel(const _Float16* __restrict__ Xt,
                     const _Float16* __restrict__ WqH, const _Float16* __restrict__ WkH,
                     const _Float16* __restrict__ WvH,
                     const float* __restrict__ bq, const float* __restrict__ bk,
                     const float* __restrict__ bv,
                     _Float16* __restrict__ Qt, _Float16* __restrict__ Kt,
                     _Float16* __restrict__ Vh) {
  __shared__ _Float16 As[2][64][40];    // [m][k] tile of W
  __shared__ _Float16 Bs[2][256][40];   // [col][k] tile of X^T

  const int tid  = threadIdx.x;
  const int lane = tid & 31, wid = tid >> 5;
  const int wm = wid & 1, wn = wid >> 1;    // 2x4 wave grid

  const int z = blockIdx.z;
  const _Float16* Wh  = (z == 0) ? WqH : (z == 1) ? WkH : WvH;
  const float*    bia = (z == 0) ? bq  : (z == 1) ? bk  : bv;
  const float mult = (z == 0) ? 0.125f : 1.0f;   // fold softmax scale 1/sqrt(64) into Q

  const int m0 = blockIdx.x * 64;
  const int g0 = blockIdx.y * 256;
  const int n  = g0 / S;
  const int s0 = g0 % S;

  // A tile: 64 rows x 4 chunk-groups = 256 chunks (1/thread)
  const int ar = tid >> 2;
  const int ag = (tid & 3) * 8;

  v8f acc[2][4] = {};

  // prologue: tile 0
  cp16(&Wh[(size_t)(m0 + ar) * C + ag], &As[0][ar][ag]);
#pragma unroll
  for (int i = 0; i < 4; ++i) {                       // B: 256 rows x 4 groups = 1024 chunks
    int ch = tid + i * 256;
    int br = ch >> 2, bg = (ch & 3) * 8;
    cp16(&Xt[((size_t)n * S + s0 + br) * C + bg], &Bs[0][br][bg]);
  }
  wait_async0();
  __syncthreads();

  for (int k = 0; k < C / 32; ++k) {
    const int cur = k & 1;
    if (k + 1 < C / 32) {
      const int kk = (k + 1) * 32;
      cp16(&Wh[(size_t)(m0 + ar) * C + kk + ag], &As[1 - cur][ar][ag]);
#pragma unroll
      for (int i = 0; i < 4; ++i) {
        int ch = tid + i * 256;
        int br = ch >> 2, bg = (ch & 3) * 8;
        cp16(&Xt[((size_t)n * S + s0 + br) * C + kk + bg], &Bs[1 - cur][br][bg]);
      }
    }
    v16h a0 = load_a_frag(&As[cur][wm * 32][0], 40, lane);
    v16h a1 = load_a_frag(&As[cur][wm * 32 + 16][0], 40, lane);
#pragma unroll
    for (int j = 0; j < 4; ++j) {
      v16h b = load_bt_frag(&Bs[cur][wn * 64 + j * 16][0], 40, lane);
      acc[0][j] = WMMA_F16(a0, b, acc[0][j]);
      acc[1][j] = WMMA_F16(a1, b, acc[1][j]);
    }
    wait_async0();
    __syncthreads();
  }

  const int half = lane >> 4;
#pragma unroll
  for (int i = 0; i < 2; ++i) {
#pragma unroll
    for (int j = 0; j < 4; ++j) {
#pragma unroll
      for (int v = 0; v < 8; ++v) {
        int m = m0 + wm * 32 + i * 16 + v + 8 * half;
        int col = s0 + wn * 64 + j * 16 + (lane & 15);
        float r = (acc[i][j][v] + bia[m]) * mult;
        if (z == 2) {
          Vh[((size_t)n * C + m) * S + col] = (_Float16)r;
        } else {
          int hh = m >> 6, d = m & 63;
          _Float16* T = (z == 0) ? Qt : Kt;
          T[(((size_t)n * NH + hh) * S + col) * HD + d] = (_Float16)r;
        }
      }
    }
  }
}

// ---------------- flash attention ----------------
// grid (S/64, NB*NH), block 128 (4 waves, 16 query rows each); double-buffered K/V tiles.
// Q A-fragments are loop-invariant and preloaded into registers.

__global__ __launch_bounds__(128)
void attn_kernel(const _Float16* __restrict__ Qt, const _Float16* __restrict__ Kt,
                 const _Float16* __restrict__ Vh, _Float16* __restrict__ Ot) {
  __shared__ _Float16 Qs[64][72];         // [s][d]
  __shared__ _Float16 Ks[2][64][72];      // [t][d]
  __shared__ _Float16 Vs[2][64][72];      // [d][t]
  __shared__ _Float16 Ps[4][16][72];      // per-wave probabilities [s][t]

  const int tid  = threadIdx.x;
  const int lane = tid & 31, wid = tid >> 5;
  const int half = lane >> 4;
  const int q0   = blockIdx.x * 64;
  const int n    = blockIdx.y / NH, h = blockIdx.y % NH;
  const size_t baseQ = ((size_t)n * NH + h) * S;      // token-major row base for Q/K
  const size_t baseV = ((size_t)n * C + h * HD) * S;  // channel-major base for V

  // 64x64 tile = 512 x 16B chunks; 128 threads -> 4 chunks each
#pragma unroll
  for (int i = 0; i < 4; ++i) {
    int ch = tid + i * 128;
    int r = ch >> 3, g = (ch & 7) * 8;
    cp16(&Qt[(baseQ + q0 + r) * HD + g], &Qs[r][g]);
    cp16(&Kt[(baseQ + 0 + r) * HD + g], &Ks[0][r][g]);
    cp16(&Vh[baseV + (size_t)r * S + 0 + g], &Vs[0][r][g]);
  }
  wait_async0();
  __syncthreads();

  // loop-invariant Q fragments (16 query rows x K-dim 64)
  v16h aq0 = load_a_frag(&Qs[wid * 16][0], 72, lane);
  v16h aq1 = load_a_frag(&Qs[wid * 16][32], 72, lane);

  float mrow[8], lrow[8];
  v8f o[4] = {};
#pragma unroll
  for (int v = 0; v < 8; ++v) { mrow[v] = -1e30f; lrow[v] = 0.f; }

  for (int it = 0; it < S / 64; ++it) {
    const int cur = it & 1;
    if (it + 1 < S / 64) {
      const int t1 = (it + 1) * 64;
#pragma unroll
      for (int i = 0; i < 4; ++i) {
        int ch = tid + i * 128;
        int r = ch >> 3, g = (ch & 7) * 8;
        cp16(&Kt[(baseQ + t1 + r) * HD + g], &Ks[1 - cur][r][g]);
        cp16(&Vh[baseV + (size_t)r * S + t1 + g], &Vs[1 - cur][r][g]);
      }
    }

    // scores: S_w(16x64) = Q_w x K^T; B[k=d][n=t] stored transposed as Ks[t][d]
    v8f sc[4] = {};
#pragma unroll
    for (int f = 0; f < 4; ++f) {
      v16h b0 = load_bt_frag(&Ks[cur][f * 16][0], 72, lane);
      sc[f] = WMMA_F16(aq0, b0, sc[f]);
      v16h b1 = load_bt_frag(&Ks[cur][f * 16][32], 72, lane);
      sc[f] = WMMA_F16(aq1, b1, sc[f]);
    }

    // online softmax; row r of a 16x16 C tile lives in one 16-lane half
    float corr[8], psum[8];
#pragma unroll
    for (int v = 0; v < 8; ++v) {
      float m_ = fmaxf(fmaxf(sc[0][v], sc[1][v]), fmaxf(sc[2][v], sc[3][v]));
      m_ = fmaxf(m_, __shfl_xor(m_, 1, 32));
      m_ = fmaxf(m_, __shfl_xor(m_, 2, 32));
      m_ = fmaxf(m_, __shfl_xor(m_, 4, 32));
      m_ = fmaxf(m_, __shfl_xor(m_, 8, 32));
      float newm = fmaxf(mrow[v], m_);
      corr[v] = __expf(mrow[v] - newm);
      mrow[v] = newm;
      psum[v] = 0.f;
    }
#pragma unroll
    for (int f = 0; f < 4; ++f) {
#pragma unroll
      for (int v = 0; v < 8; ++v) {
        float p = __expf(sc[f][v] - mrow[v]);
        psum[v] += p;
        Ps[wid][v + 8 * half][f * 16 + (lane & 15)] = (_Float16)p;
        o[f][v] *= corr[v];
      }
    }
#pragma unroll
    for (int v = 0; v < 8; ++v) {
      float ss = psum[v];
      ss += __shfl_xor(ss, 1, 32);
      ss += __shfl_xor(ss, 2, 32);
      ss += __shfl_xor(ss, 4, 32);
      ss += __shfl_xor(ss, 8, 32);
      lrow[v] = lrow[v] * corr[v] + ss;
    }

    // per-wave LDS RAW: P stores -> A-frag loads
    asm volatile("s_wait_dscnt 0" ::: "memory");

    // O += P(16x64) x V^T; B[k=t][n=d] stored transposed as Vs[d][t]
#pragma unroll
    for (int kk = 0; kk < 2; ++kk) {
      v16h ap = load_a_frag(&Ps[wid][0][kk * 32], 72, lane);
#pragma unroll
      for (int f = 0; f < 4; ++f) {
        v16h bvf = load_bt_frag(&Vs[cur][f * 16][kk * 32], 72, lane);
        o[f] = WMMA_F16(ap, bvf, o[f]);
      }
    }

    wait_async0();
    __syncthreads();
  }

  // normalize; store token-major Ot [n][s][c]
#pragma unroll
  for (int f = 0; f < 4; ++f) {
#pragma unroll
    for (int v = 0; v < 8; ++v) {
      int d = f * 16 + (lane & 15);
      int s = q0 + wid * 16 + v + 8 * half;
      Ot[((size_t)n * S + s) * C + h * HD + d] = (_Float16)(o[f][v] / lrow[v]);
    }
  }
}

// ---------------- output projection + residual: y = gamma*(Wo*O + bo) + x ----------------
// grid (C/64, NB*S/256), block 256; same tiling as qkv

__global__ __launch_bounds__(256)
void out_proj_kernel(const _Float16* __restrict__ Ot, const _Float16* __restrict__ WoH,
                     const float* __restrict__ bo, const float* __restrict__ gamma,
                     const float* __restrict__ x, float* __restrict__ y) {
  __shared__ _Float16 As[2][64][40];
  __shared__ _Float16 Bs[2][256][40];

  const int tid  = threadIdx.x;
  const int lane = tid & 31, wid = tid >> 5;
  const int wm = wid & 1, wn = wid >> 1;

  const int m0 = blockIdx.x * 64;
  const int g0 = blockIdx.y * 256;
  const int n  = g0 / S;
  const int s0 = g0 % S;

  const int ar = tid >> 2;
  const int ag = (tid & 3) * 8;

  v8f acc[2][4] = {};

  cp16(&WoH[(size_t)(m0 + ar) * C + ag], &As[0][ar][ag]);
#pragma unroll
  for (int i = 0; i < 4; ++i) {
    int ch = tid + i * 256;
    int br = ch >> 2, bg = (ch & 3) * 8;
    cp16(&Ot[((size_t)n * S + s0 + br) * C + bg], &Bs[0][br][bg]);
  }
  wait_async0();
  __syncthreads();

  for (int k = 0; k < C / 32; ++k) {
    const int cur = k & 1;
    if (k + 1 < C / 32) {
      const int kk = (k + 1) * 32;
      cp16(&WoH[(size_t)(m0 + ar) * C + kk + ag], &As[1 - cur][ar][ag]);
#pragma unroll
      for (int i = 0; i < 4; ++i) {
        int ch = tid + i * 256;
        int br = ch >> 2, bg = (ch & 3) * 8;
        cp16(&Ot[((size_t)n * S + s0 + br) * C + kk + bg], &Bs[1 - cur][br][bg]);
      }
    }
    v16h a0 = load_a_frag(&As[cur][wm * 32][0], 40, lane);
    v16h a1 = load_a_frag(&As[cur][wm * 32 + 16][0], 40, lane);
#pragma unroll
    for (int j = 0; j < 4; ++j) {
      v16h b = load_bt_frag(&Bs[cur][wn * 64 + j * 16][0], 40, lane);
      acc[0][j] = WMMA_F16(a0, b, acc[0][j]);
      acc[1][j] = WMMA_F16(a1, b, acc[1][j]);
    }
    wait_async0();
    __syncthreads();
  }

  const float g = gamma[0];
  const int half = lane >> 4;
#pragma unroll
  for (int i = 0; i < 2; ++i) {
#pragma unroll
    for (int j = 0; j < 4; ++j) {
#pragma unroll
      for (int v = 0; v < 8; ++v) {
        int m = m0 + wm * 32 + i * 16 + v + 8 * half;
        size_t off = ((size_t)n * C + m) * S + s0 + wn * 64 + j * 16 + (lane & 15);
        y[off] = g * (acc[i][j][v] + bo[m]) + x[off];
      }
    }
  }
}

// ---------------- launch ----------------

extern "C" void kernel_launch(void* const* d_in, const int* in_sizes, int n_in,
                              void* d_out, int out_size, void* d_ws, size_t ws_size,
                              hipStream_t stream) {
  const float* x     = (const float*)d_in[0];
  const float* Wq    = (const float*)d_in[1];
  const float* bq    = (const float*)d_in[2];
  const float* Wk    = (const float*)d_in[3];
  const float* bk    = (const float*)d_in[4];
  const float* Wv    = (const float*)d_in[5];
  const float* bv    = (const float*)d_in[6];
  const float* Wo    = (const float*)d_in[7];
  const float* bo    = (const float*)d_in[8];
  const float* gamma = (const float*)d_in[9];

  char* ws = (char*)d_ws;
  const size_t MB = 1024ull * 1024ull;
  _Float16* Xt  = (_Float16*)(ws + 0 * MB);    // [n][s][c] f16, 8 MB
  _Float16* Qt  = (_Float16*)(ws + 8 * MB);    // [n][h][s][d]
  _Float16* Kt  = (_Float16*)(ws + 16 * MB);   // [n][h][t][d]
  _Float16* Vh  = (_Float16*)(ws + 24 * MB);   // [n][c][t]
  _Float16* Ot  = (_Float16*)(ws + 32 * MB);   // [n][s][c]
  _Float16* WqH = (_Float16*)(ws + 40 * MB);   // each 512 KB
  _Float16* WkH = (_Float16*)(ws + 40 * MB + 512 * 1024);
  _Float16* WvH = (_Float16*)(ws + 41 * MB);
  _Float16* WoH = (_Float16*)(ws + 41 * MB + 512 * 1024);

  const int nx = NB * C * S;
  const int nw = C * C;
  xpose_f16_kernel<<<(nx + 255) / 256, 256, 0, stream>>>(x, Xt);
  f32_to_f16_kernel<<<(nw + 255) / 256, 256, 0, stream>>>(Wq, WqH, nw);
  f32_to_f16_kernel<<<(nw + 255) / 256, 256, 0, stream>>>(Wk, WkH, nw);
  f32_to_f16_kernel<<<(nw + 255) / 256, 256, 0, stream>>>(Wv, WvH, nw);
  f32_to_f16_kernel<<<(nw + 255) / 256, 256, 0, stream>>>(Wo, WoH, nw);

  dim3 gqkv(C / 64, NB * S / 256, 3);
  qkv_gemm_kernel<<<gqkv, 256, 0, stream>>>(Xt, WqH, WkH, WvH, bq, bk, bv, Qt, Kt, Vh);

  dim3 gat(S / 64, NB * NH);
  attn_kernel<<<gat, 128, 0, stream>>>(Qt, Kt, Vh, Ot);

  dim3 gout(C / 64, NB * S / 256);
  out_proj_kernel<<<gout, 256, 0, stream>>>(Ot, WoH, bo, gamma, x, (float*)d_out);
}
